// SwinBlock_27212912787509
// MI455X (gfx1250) — matrix operand order repaired and checked by reading, not compiled, not run on hardware
//
#include <hip/hip_runtime.h>
#include <math.h>

// ---------------------------------------------------------------------------
// Swin block: roll+LN1+partition -> QKV -> windowed MHSA -> proj+reverse+res
//             -> fused (LN2 -> fc1 -> GELU -> fc2 -> res)
// All GEMMs via v_wmma_f32_16x16x32_f16 (CDNA5), fp32 accumulation.
// ---------------------------------------------------------------------------

typedef __attribute__((ext_vector_type(16))) _Float16 v16h;
typedef __attribute__((ext_vector_type(8)))  _Float16 v8h;
typedef __attribute__((ext_vector_type(8)))  float    v8f;

#define DEVI static __device__ __forceinline__

// A fragment (16x32 f16, M x K), source row-major with leading dim `ld` halfs,
// K contiguous. Per ISA: lane L holds row M=L%16; halfs 0..7 at K=8*(L/16),
// halfs 8..15 at K=16+8*(L/16).  Two 16B contiguous loads.
DEVI v16h load_afrag(const _Float16* tile, int ld, int lane) {
    int m = lane & 15, g = lane >> 4;
    const _Float16* p = tile + m * ld + g * 8;
    v8h lo = *(const v8h*)p;
    v8h hi = *(const v8h*)(p + 16);
    v16h r;
#pragma unroll
    for (int i = 0; i < 8; ++i) { r[i] = lo[i]; r[i + 8] = hi[i]; }
    return r;
}

// B fragment (32x16 f16, K x N) sourced from row-major (N x K) storage with
// leading dim `ld` halfs (K contiguous). Lane L: col N=L%16; halfs 0..7 at
// K=8*(L/16), halfs 8..15 at K=16+8*(L/16).  Two 16B contiguous loads.
DEVI v16h load_bfrag(const _Float16* tile, int ld, int lane) {
    int n = lane & 15, g = lane >> 4;
    const _Float16* p = tile + n * ld + g * 8;
    v8h lo = *(const v8h*)p;
    v8h hi = *(const v8h*)(p + 16);
    v16h r;
#pragma unroll
    for (int i = 0; i < 8; ++i) { r[i] = lo[i]; r[i + 8] = hi[i]; }
    return r;
}

DEVI v8f wmma_f16(v16h a, v16h b, v8f c) {
    return __builtin_amdgcn_wmma_f32_16x16x32_f16(false, a, false, b,
                                                  (short)0, c, false, false);
}

// ---------------------------------------------------------------------------
// K0: fp32 -> fp16 weight conversion
// ---------------------------------------------------------------------------
__global__ void cvt_f32_f16(const float* __restrict__ src,
                            _Float16* __restrict__ dst, int n) {
    int i = blockIdx.x * blockDim.x + threadIdx.x;
    if (i < n) dst[i] = (_Float16)src[i];
}

// ---------------------------------------------------------------------------
// K1: roll(-8,-8) + LayerNorm1 + window partition -> win f16 (131072 x 128)
// One wave per token row; wave32 shfl reductions.
// Row r = win*64 + t ; win = b*64 + hI*8 + wI ; t = sy*8 + sx
// source pixel: y=(sy*8+hI+8)%64, x=(sx*8+wI+8)%64  (roll folded in)
// ---------------------------------------------------------------------------
__global__ void ln1_partition(const float* __restrict__ x,
                              const float* __restrict__ w,
                              const float* __restrict__ b,
                              _Float16* __restrict__ win) {
    int lane = threadIdx.x & 31;
    int row  = blockIdx.x * (blockDim.x >> 5) + (threadIdx.x >> 5);
    int wi = row >> 6, t = row & 63;
    int bb = wi >> 6, hw = wi & 63;
    int hI = hw >> 3, wI = hw & 7;
    int sy = t >> 3,  sx = t & 7;
    int ys = (sy * 8 + hI + 8) & 63;
    int xs = (sx * 8 + wI + 8) & 63;
    const float* src = x + ((size_t)bb * 4096 + ys * 64 + xs) * 128;

    float v[4], s = 0.f, s2 = 0.f;
#pragma unroll
    for (int i = 0; i < 4; ++i) {
        v[i] = src[lane * 4 + i];
        s += v[i]; s2 += v[i] * v[i];
    }
#pragma unroll
    for (int m = 16; m >= 1; m >>= 1) {
        s  += __shfl_xor(s,  m, 32);
        s2 += __shfl_xor(s2, m, 32);
    }
    float mean = s * (1.f / 128.f);
    float var  = s2 * (1.f / 128.f) - mean * mean;
    float rstd = rsqrtf(var + 1e-5f);
    _Float16* dst = win + (size_t)row * 128 + lane * 4;
#pragma unroll
    for (int i = 0; i < 4; ++i) {
        int c = lane * 4 + i;
        dst[i] = (_Float16)(((v[i] - mean) * rstd) * w[c] + b[c]);
    }
}

// ---------------------------------------------------------------------------
// K2: QKV GEMM. M=131072, K=128, N=384. One wave computes a 16x64 tile.
// Output de-interleaved: n = head*96 + which*32 + d  ->  q/k/v buffers laid
// out as (win, head, tok, d) with d contiguous (clean WMMA fragments later).
// ---------------------------------------------------------------------------
__global__ void qkv_gemm(const _Float16* __restrict__ A,
                         const _Float16* __restrict__ Bw,
                         const float* __restrict__ bias,
                         _Float16* __restrict__ q,
                         _Float16* __restrict__ k,
                         _Float16* __restrict__ v) {
    int lane = threadIdx.x;
    int m0 = blockIdx.x * 16;
    int nb = blockIdx.y * 64;
    v8f zero = {};
    v8f acc[4] = {zero, zero, zero, zero};
    for (int kk = 0; kk < 128; kk += 32) {
        v16h a = load_afrag(A + (size_t)m0 * 128 + kk, 128, lane);
#pragma unroll
        for (int j = 0; j < 4; ++j) {
            v16h bf = load_bfrag(Bw + (size_t)(nb + j * 16) * 128 + kk, 128, lane);
            acc[j] = wmma_f16(a, bf, acc[j]);
        }
    }
    int wi = m0 >> 6;                 // 16-row tile never crosses a window
    int n = lane & 15, g = lane >> 4;
#pragma unroll
    for (int j = 0; j < 4; ++j) {
        int n0 = nb + j * 16;
        int head = n0 / 96, r = n0 % 96;
        int which = r / 32, d0 = r % 32;
        _Float16* dst = (which == 0) ? q : (which == 1) ? k : v;
        float bv = bias[n0 + n];
        int col = d0 + n;
#pragma unroll
        for (int i = 0; i < 8; ++i) {
            int tok = (m0 + i + 8 * g) & 63;
            dst[(((size_t)wi * 4 + head) * 64 + tok) * 32 + col] =
                (_Float16)(acc[j][i] + bv);
        }
    }
}

// ---------------------------------------------------------------------------
// K3: attention per (window, head): energy = q k^T * scale + 1 -> softmax
//     -> out = attn @ v.  Block = 4 waves (128 threads). All WMMA.
// ---------------------------------------------------------------------------
__global__ void attn_kernel(const _Float16* __restrict__ q,
                            const _Float16* __restrict__ k,
                            const _Float16* __restrict__ v,
                            _Float16* __restrict__ ao) {
    __shared__ __align__(16) _Float16 vt[32][72];   // V transposed (d, tok)
    __shared__ float e[64][64];                     // energies fp32
    __shared__ __align__(16) _Float16 p[64][72];    // softmax probs f16

    int wh = blockIdx.x;            // window*4 + head
    int head = wh & 3, win = wh >> 2;
    int tid = threadIdx.x, lane = tid & 31, wv = tid >> 5;
    const _Float16* qb = q + (size_t)wh * 64 * 32;
    const _Float16* kb = k + (size_t)wh * 64 * 32;
    const _Float16* vb = v + (size_t)wh * 64 * 32;

    for (int i = tid; i < 64 * 32; i += 128) {      // transpose V into LDS
        int tok = i >> 5, d = i & 31;
        vt[d][tok] = vb[i];
    }

    {   // energy: wave wv does rows [wv*16, wv*16+16), all 64 cols
        int m0 = wv * 16;
        v16h a = load_afrag(qb + m0 * 32, 32, lane);
        int n = lane & 15, g = lane >> 4;
#pragma unroll
        for (int j = 0; j < 4; ++j) {
            v8f c = {};
            v16h bf = load_bfrag(kb + j * 16 * 32, 32, lane);
            c = wmma_f16(a, bf, c);
#pragma unroll
            for (int i = 0; i < 8; ++i)
                e[m0 + i + 8 * g][j * 16 + n] =
                    c[i] * 0.17677669529663687f + 1.0f;   // scale=HD^-0.5, +mask
        }
    }
    __syncthreads();

    if (tid < 64) {                 // softmax, one row per thread
        float mx = -1e30f;
        for (int j = 0; j < 64; ++j) mx = fmaxf(mx, e[tid][j]);
        float s = 0.f;
        for (int j = 0; j < 64; ++j) s += __expf(e[tid][j] - mx);
        float inv = 1.0f / s;
        for (int j = 0; j < 64; ++j)
            p[tid][j] = (_Float16)(__expf(e[tid][j] - mx) * inv);
    }
    __syncthreads();

    {   // out = p @ v : rows [wv*16, +16), 32 cols, K=64
        int m0 = wv * 16;
        int n = lane & 15, g = lane >> 4;
#pragma unroll
        for (int j = 0; j < 2; ++j) {
            v8f c = {};
#pragma unroll
            for (int kk = 0; kk < 64; kk += 32) {
                v16h a  = load_afrag(&p[m0][kk], 72, lane);
                v16h bf = load_bfrag(&vt[j * 16][kk], 72, lane);
                c = wmma_f16(a, bf, c);
            }
#pragma unroll
            for (int i = 0; i < 8; ++i) {
                int tok = m0 + i + 8 * g;
                ao[((size_t)win * 64 + tok) * 128 + head * 32 + j * 16 + n] =
                    (_Float16)c[i];
            }
        }
    }
}

// ---------------------------------------------------------------------------
// K4: proj GEMM + window-reverse + roll(+8,+8) + residual -> msa_out (fp32)
// Scatter target equals the K1 gather source (roll/partition are inverses).
// ---------------------------------------------------------------------------
__global__ void proj_gemm(const _Float16* __restrict__ A,
                          const _Float16* __restrict__ Bw,
                          const float* __restrict__ bias,
                          const float* __restrict__ xin,
                          float* __restrict__ out) {
    int lane = threadIdx.x;
    int m0 = blockIdx.x * 16;
    int nb = blockIdx.y * 64;
    v8f zero = {};
    v8f acc[4] = {zero, zero, zero, zero};
    for (int kk = 0; kk < 128; kk += 32) {
        v16h a = load_afrag(A + (size_t)m0 * 128 + kk, 128, lane);
#pragma unroll
        for (int j = 0; j < 4; ++j) {
            v16h bf = load_bfrag(Bw + (size_t)(nb + j * 16) * 128 + kk, 128, lane);
            acc[j] = wmma_f16(a, bf, acc[j]);
        }
    }
    int wi = m0 >> 6;
    int bb = wi >> 6, hw = wi & 63;
    int hI = hw >> 3, wI = hw & 7;
    int n = lane & 15, g = lane >> 4;
#pragma unroll
    for (int j = 0; j < 4; ++j) {
        float bv = bias[nb + j * 16 + n];
#pragma unroll
        for (int i = 0; i < 8; ++i) {
            int t = (m0 + i + 8 * g) & 63;
            int sy = t >> 3, sx = t & 7;
            int ys = (sy * 8 + hI + 8) & 63;
            int xs = (sx * 8 + wI + 8) & 63;
            size_t idx = ((size_t)bb * 4096 + ys * 64 + xs) * 128 + nb + j * 16 + n;
            out[idx] = xin[idx] + acc[j][i] + bv;
        }
    }
}

// ---------------------------------------------------------------------------
// K5: fused MLP per 16-row tile: LN2 -> fc1(128->512) -> exact GELU -> LDS
//     -> fc2(512->128) -> d_out += result.  Block = 4 waves.
// ---------------------------------------------------------------------------
__global__ void mlp_fused(float* __restrict__ io,
                          const float* __restrict__ ln2w,
                          const float* __restrict__ ln2b,
                          const _Float16* __restrict__ w1,
                          const float* __restrict__ b1,
                          const _Float16* __restrict__ w2,
                          const float* __restrict__ b2) {
    __shared__ __align__(16) _Float16 xln[16][128];
    __shared__ __align__(16) _Float16 hid[16][520];
    __shared__ float red[16][16];
    __shared__ float mrs[16][2];

    int tid = threadIdx.x, lane = tid & 31, wv = tid >> 5;
    size_t row0 = (size_t)blockIdx.x * 16;

    // LayerNorm2: 8 threads per row, 16 channels each
    int r = tid >> 3, c0 = (tid & 7) * 16;
    const float* src = io + (row0 + r) * 128 + c0;
    float vals[16], s = 0.f, s2 = 0.f;
#pragma unroll
    for (int i = 0; i < 16; ++i) {
        vals[i] = src[i];
        s += vals[i]; s2 += vals[i] * vals[i];
    }
    red[r][tid & 7] = s;
    red[r][8 + (tid & 7)] = s2;
    __syncthreads();
    if (tid < 16) {
        float ss = 0.f, ss2 = 0.f;
#pragma unroll
        for (int i = 0; i < 8; ++i) { ss += red[tid][i]; ss2 += red[tid][8 + i]; }
        float mean = ss * (1.f / 128.f);
        float var  = ss2 * (1.f / 128.f) - mean * mean;
        mrs[tid][0] = mean;
        mrs[tid][1] = rsqrtf(var + 1e-5f);
    }
    __syncthreads();
    {
        float mean = mrs[r][0], rstd = mrs[r][1];
#pragma unroll
        for (int i = 0; i < 16; ++i) {
            int c = c0 + i;
            xln[r][c] = (_Float16)((vals[i] - mean) * rstd * ln2w[c] + ln2b[c]);
        }
    }
    __syncthreads();

    // fc1 + GELU: wave wv owns hidden cols [wv*128, +128) -> 8 tiles, K=128
    {
        v16h a[4];
#pragma unroll
        for (int kk = 0; kk < 4; ++kk)
            a[kk] = load_afrag(&xln[0][kk * 32], 128, lane);
        int n = lane & 15, g = lane >> 4;
#pragma unroll
        for (int jt = 0; jt < 8; ++jt) {
            int n0 = wv * 128 + jt * 16;
            v8f c = {};
#pragma unroll
            for (int kk = 0; kk < 4; ++kk) {
                v16h bf = load_bfrag(w1 + (size_t)n0 * 128 + kk * 32, 128, lane);
                c = wmma_f16(a[kk], bf, c);
            }
            float bv = b1[n0 + n];
#pragma unroll
            for (int i = 0; i < 8; ++i) {
                float h = c[i] + bv;
                float gel = 0.5f * h * (1.0f + erff(h * 0.70710678118654752f));
                hid[i + 8 * g][n0 + n] = (_Float16)gel;
            }
        }
    }
    __syncthreads();

    // fc2: wave wv owns out cols [wv*32, +32) -> 2 tiles, K=512
    {
        int n = lane & 15, g = lane >> 4;
#pragma unroll
        for (int jt = 0; jt < 2; ++jt) {
            int n0 = wv * 32 + jt * 16;
            v8f c = {};
            for (int kk = 0; kk < 512; kk += 32) {
                v16h a  = load_afrag(&hid[0][kk], 520, lane);
                v16h bf = load_bfrag(w2 + (size_t)n0 * 512 + kk, 512, lane);
                c = wmma_f16(a, bf, c);
            }
            float bv = b2[n0 + n];
#pragma unroll
            for (int i = 0; i < 8; ++i) {
                size_t rr = row0 + i + 8 * g;
                io[rr * 128 + n0 + n] += c[i] + bv;
            }
        }
    }
}

// ---------------------------------------------------------------------------
// Host launcher
// ---------------------------------------------------------------------------
extern "C" void kernel_launch(void* const* d_in, const int* in_sizes, int n_in,
                              void* d_out, int out_size, void* d_ws, size_t ws_size,
                              hipStream_t stream) {
    (void)in_sizes; (void)n_in; (void)out_size; (void)ws_size;
    const float* x     = (const float*)d_in[0];
    const float* ln1w  = (const float*)d_in[1];
    const float* ln1b  = (const float*)d_in[2];
    const float* qkvw  = (const float*)d_in[3];
    const float* qkvb  = (const float*)d_in[4];
    const float* projw = (const float*)d_in[5];
    const float* projb = (const float*)d_in[6];
    const float* ln2w  = (const float*)d_in[7];
    const float* ln2b  = (const float*)d_in[8];
    const float* fc1w  = (const float*)d_in[9];
    const float* fc1b  = (const float*)d_in[10];
    const float* fc2w  = (const float*)d_in[11];
    const float* fc2b  = (const float*)d_in[12];
    float* out = (float*)d_out;

    char* ws = (char*)d_ws;
    const size_t ACT = (size_t)131072 * 128 * sizeof(_Float16); // 33.5 MB each
    _Float16* qkvw_h = (_Float16*)(ws);                         // 98304 B
    _Float16* projw_h = (_Float16*)(ws + 98304);                // 32768 B
    _Float16* fc1w_h  = (_Float16*)(ws + 131072);               // 131072 B
    _Float16* fc2w_h  = (_Float16*)(ws + 262144);               // 131072 B
    _Float16* win_h   = (_Float16*)(ws + 393216);
    _Float16* q_h     = (_Float16*)(ws + 393216 + 1 * ACT);
    _Float16* k_h     = (_Float16*)(ws + 393216 + 2 * ACT);
    _Float16* v_h     = (_Float16*)(ws + 393216 + 3 * ACT);
    _Float16* ao_h    = (_Float16*)(ws + 393216 + 4 * ACT);

    // K0: weight conversion
    cvt_f32_f16<<<(49152 + 255) / 256, 256, 0, stream>>>(qkvw, qkvw_h, 49152);
    cvt_f32_f16<<<(16384 + 255) / 256, 256, 0, stream>>>(projw, projw_h, 16384);
    cvt_f32_f16<<<(65536 + 255) / 256, 256, 0, stream>>>(fc1w, fc1w_h, 65536);
    cvt_f32_f16<<<(65536 + 255) / 256, 256, 0, stream>>>(fc2w, fc2w_h, 65536);

    // K1: roll + LN1 + window partition (131072 rows, 4 rows/block)
    ln1_partition<<<32768, 128, 0, stream>>>(x, ln1w, ln1b, win_h);

    // K2: QKV GEMM  (8192 M-tiles x 6 N-groups of 64)
    qkv_gemm<<<dim3(8192, 6), 32, 0, stream>>>(win_h, qkvw_h, qkvb, q_h, k_h, v_h);

    // K3: attention, one block per (window, head)
    attn_kernel<<<2048 * 4, 128, 0, stream>>>(q_h, k_h, v_h, ao_h);

    // K4: proj GEMM + reverse + residual -> d_out = msa_out
    proj_gemm<<<dim3(8192, 2), 32, 0, stream>>>(ao_h, projw_h, projb, x, out);

    // K5: fused LN2 + fc1 + GELU + fc2 + residual, in-place on d_out
    mlp_fused<<<8192, 128, 0, stream>>>(out, ln2w, ln2b, fc1w_h, fc1b, fc2w_h, fc2b);
}
